// PerElementModule_41154376630635
// MI455X (gfx1250) — compile-verified, matrix operand order checked by reading, not run on hardware
//
#include <hip/hip_runtime.h>
#include <hip/hip_bf16.h>

#define N_ELEM 4
#define DESC 128
#define HID 128
#define BPE 256                 // blocks per expert in the GEMM kernel
#define WAVES_PER_BLOCK 4
#define THREADS (WAVES_PER_BLOCK * 32)
#define TSTRIDE (BPE * WAVES_PER_BLOCK)

typedef __attribute__((ext_vector_type(16))) __bf16 v16bf;
typedef __attribute__((ext_vector_type(8)))  __bf16 v8bf;
typedef __attribute__((ext_vector_type(8)))  float  v8f;
typedef __attribute__((ext_vector_type(4)))  float  v4f;

// ws layout (unsigned int): [0..3]=count, [4..7]=base, [8..11]=cursor, [16..]=idx[N]

__global__ void zero_ws_kernel(unsigned int* ws) {
    if (threadIdx.x < 12) ws[threadIdx.x] = 0u;
}

__global__ void count_kernel(const int* __restrict__ element, int n,
                             unsigned int* __restrict__ ws) {
    __shared__ unsigned int lc[N_ELEM];
    if (threadIdx.x < N_ELEM) lc[threadIdx.x] = 0u;
    __syncthreads();
    int i = blockIdx.x * blockDim.x + threadIdx.x;
    if (i < n) atomicAdd(&lc[element[i]], 1u);
    __syncthreads();
    if (threadIdx.x < N_ELEM) atomicAdd(&ws[threadIdx.x], lc[threadIdx.x]);
}

__global__ void prefix_kernel(unsigned int* ws) {
    unsigned int b = 0;
    for (int e = 0; e < N_ELEM; ++e) {
        ws[4 + e] = b;
        ws[8 + e] = b;
        b += ws[e];
    }
}

__global__ void scatter_kernel(const int* __restrict__ element, int n,
                               unsigned int* __restrict__ ws,
                               unsigned int* __restrict__ idx) {
    __shared__ unsigned int lc[N_ELEM];
    __shared__ unsigned int lbase[N_ELEM];
    if (threadIdx.x < N_ELEM) lc[threadIdx.x] = 0u;
    __syncthreads();
    int i = blockIdx.x * blockDim.x + threadIdx.x;
    int e = 0; unsigned int lpos = 0;
    if (i < n) { e = element[i]; lpos = atomicAdd(&lc[e], 1u); }
    __syncthreads();
    if (threadIdx.x < N_ELEM)
        lbase[threadIdx.x] = atomicAdd(&ws[8 + threadIdx.x], lc[threadIdx.x]);
    __syncthreads();
    if (i < n) idx[lbase[e] + lpos] = (unsigned int)i;
}

// One wave handles a 16-atom tile of a single expert:
//   C[16x16 per N-tile] = A[16 atoms x 128 desc, bf16] * W1^T-slice, f32 accum
// then bias+relu+dot(W2) folded in as a per-lane partial, cross-lane reduced.
__global__ __launch_bounds__(THREADS) void moe_wmma_kernel(
    const float* __restrict__ desc,
    const float* __restrict__ W1,   // [4][128][128] (k rows, h columns)
    const float* __restrict__ b1,   // [4][128]
    const float* __restrict__ W2,   // [4][128]
    const float* __restrict__ b2,   // [4]
    const unsigned int* __restrict__ ws,
    const unsigned int* __restrict__ idx,
    float* __restrict__ out)
{
    const int e    = blockIdx.x / BPE;
    const int slot = blockIdx.x % BPE;
    const unsigned int cnt  = ws[e];
    const unsigned int base = ws[4 + e];
    if (cnt == 0) return;

    // LDS: W1 transposed to [h][k] in bf16, rows padded to 136 elems (272B)
    // so the 16 B-fragment lanes (stride 272B -> 4 banks apart) are conflict-free.
    constexpr int ROW = DESC + 8;
    __shared__ __bf16 w1t[HID * ROW];
    __shared__ float  b1s[HID];
    __shared__ float  w2s[HID];

    for (int i = threadIdx.x; i < DESC * HID; i += THREADS) {
        int k = i >> 7;            // 0..127 (desc dim)
        int h = i & (HID - 1);     // 0..127 (hidden)
        w1t[h * ROW + k] = (__bf16)W1[(size_t)e * DESC * HID + i];
    }
    for (int i = threadIdx.x; i < HID; i += THREADS) {
        b1s[i] = b1[e * HID + i];
        w2s[i] = W2[e * HID + i];
    }
    __syncthreads();

    const int lane = threadIdx.x & 31;
    const int wave = threadIdx.x >> 5;
    const int row  = lane & 15;    // A-matrix row owned by this lane
    const int half = lane >> 4;    // K-half split per ISA A/B fragment layout
    const float b2v = b2[e];

    // B fragments for all 8 N-tiles live in registers for the whole loop
    // (invariant across tiles; compiler keeps them resident).
    v16bf Bf[8][4];
#pragma unroll
    for (int j = 0; j < 8; ++j) {
        const int h = j * 16 + row;
#pragma unroll
        for (int c = 0; c < 4; ++c) {
            const int k0 = c * 32 + half * 8;
            const v8bf blo = *(const v8bf*)&w1t[h * ROW + k0];
            const v8bf bhi = *(const v8bf*)&w1t[h * ROW + k0 + 16];
            Bf[j][c] = __builtin_shufflevector(
                blo, bhi, 0,1,2,3,4,5,6,7,8,9,10,11,12,13,14,15);
        }
    }

    const unsigned int ntile = (cnt + 15u) >> 4;
    for (unsigned int t = (unsigned int)(slot * WAVES_PER_BLOCK + wave);
         t < ntile; t += TSTRIDE) {

        const unsigned int gm   = t * 16u + (unsigned int)row;
        const unsigned int gi   = gm < cnt ? gm : cnt - 1u;   // clamp tail (EXEC stays all-1s)
        const unsigned int atom = idx[base + gi];
        const float* dr = desc + (size_t)atom * DESC;

        // Prefetch next tile's descriptor row: half 0 covers cachelines 0-1,
        // half 1 covers cachelines 2-3 of its row (global_prefetch_b8 path).
        {
            const unsigned int tn = t + TSTRIDE;
            if (tn < ntile) {
                unsigned int gmn = tn * 16u + (unsigned int)row;
                unsigned int gin = gmn < cnt ? gmn : cnt - 1u;
                const float* drn = desc + (size_t)idx[base + gin] * DESC;
                __builtin_prefetch(drn + half * 64, 0, 0);
                __builtin_prefetch(drn + half * 64 + 32, 0, 0);
            }
        }

        // A fragments: 4 K-chunks of 32. Lane layout (16-bit A 16x32):
        // elems 0..7 = K[k0..k0+7], elems 8..15 = K[k0+16..k0+23], k0=c*32+half*8.
        v16bf A[4];
#pragma unroll
        for (int c = 0; c < 4; ++c) {
            const int k0 = c * 32 + half * 8;
            const v4f lo0 = *(const v4f*)(dr + k0);
            const v4f lo1 = *(const v4f*)(dr + k0 + 4);
            const v4f hi0 = *(const v4f*)(dr + k0 + 16);
            const v4f hi1 = *(const v4f*)(dr + k0 + 20);
#pragma unroll
            for (int q = 0; q < 4; ++q) {
                A[c][q]      = (__bf16)lo0[q];
                A[c][4 + q]  = (__bf16)lo1[q];
                A[c][8 + q]  = (__bf16)hi0[q];
                A[c][12 + q] = (__bf16)hi1[q];
            }
        }

        float p[8];
#pragma unroll
        for (int r = 0; r < 8; ++r) p[r] = 0.0f;

        // Two live accumulators per step so the j+1 WMMA chain overlaps the
        // j epilogue (fills the WMMA->VALU hazard slots; WMMA co-executes
        // with VALU as a TRANS-class op).
#pragma unroll
        for (int j = 0; j < 8; j += 2) {
            v8f acc0 = {};
            v8f acc1 = {};
#pragma unroll
            for (int c = 0; c < 4; ++c)
                acc0 = __builtin_amdgcn_wmma_f32_16x16x32_bf16(
                    false, A[c], false, Bf[j][c], (short)0, acc0, false, false);
#pragma unroll
            for (int c = 0; c < 4; ++c)
                acc1 = __builtin_amdgcn_wmma_f32_16x16x32_bf16(
                    false, A[c], false, Bf[j + 1][c], (short)0, acc1, false, false);

            // Epilogue: lane holds C[m][n] for n = h, m = r + half*8 (VGPR r).
            const int h0 = j * 16 + row;
            const int h1 = (j + 1) * 16 + row;
            const float b1v0 = b1s[h0], w2v0 = w2s[h0];
            const float b1v1 = b1s[h1], w2v1 = w2s[h1];
#pragma unroll
            for (int r = 0; r < 8; ++r) {
                float hv0 = acc0[r] + b1v0;
                hv0 = hv0 > 0.0f ? hv0 : 0.0f;
                p[r] += hv0 * w2v0;
                float hv1 = acc1[r] + b1v1;
                hv1 = hv1 > 0.0f ? hv1 : 0.0f;
                p[r] += hv1 * w2v1;
            }
        }

        // Reduce layer-2 partials across the 16 lanes sharing each output row.
#pragma unroll
        for (int r = 0; r < 8; ++r) {
#pragma unroll
            for (int m = 1; m < 16; m <<= 1)
                p[r] += __shfl_xor(p[r], m, 32);
        }

        // All 16 lanes of a half now hold all 8 row sums; spread the stores
        // across 8 lanes per half (lane row<8 stores row half*8+row).
        if (row < 8) {
            float yv = p[0];
#pragma unroll
            for (int r = 1; r < 8; ++r)
                if (row == r) yv = p[r];
            const unsigned int om = t * 16u + (unsigned int)(half * 8 + row);
            if (om < cnt) out[idx[base + om]] = yv + b2v;
        }
    }
}

extern "C" void kernel_launch(void* const* d_in, const int* in_sizes, int n_in,
                              void* d_out, int out_size, void* d_ws, size_t ws_size,
                              hipStream_t stream) {
    const int*   element = (const int*)d_in[0];
    const float* desc    = (const float*)d_in[1];
    const float* W1      = (const float*)d_in[2];
    const float* b1      = (const float*)d_in[3];
    const float* W2      = (const float*)d_in[4];
    const float* b2      = (const float*)d_in[5];
    float* out = (float*)d_out;
    const int n = in_sizes[0];

    unsigned int* ws  = (unsigned int*)d_ws;
    unsigned int* idx = ws + 16;   // 64B in; needs 64 + 4*n bytes of scratch

    zero_ws_kernel<<<1, 32, 0, stream>>>(ws);
    const int cb = (n + 255) / 256;
    count_kernel<<<cb, 256, 0, stream>>>(element, n, ws);
    prefix_kernel<<<1, 1, 0, stream>>>(ws);
    scatter_kernel<<<cb, 256, 0, stream>>>(element, n, ws, idx);
    moe_wmma_kernel<<<N_ELEM * BPE, THREADS, 0, stream>>>(
        desc, W1, b1, W2, b2, ws, idx, out);
}
